// GQA_13245679141285
// MI455X (gfx1250) — matrix operand order repaired and checked by reading, not compiled, not run on hardware
//
#include <hip/hip_runtime.h>
#include <hip/hip_bf16.h>

// ---------------- problem constants (match reference) ----------------
#define BB   2
#define SS   2048
#define DD   2048
#define HH   16
#define KVH  4
#define HD   128
#define GG   (HH / KVH)          // 4
#define NKV  (2 * KVH * HD)      // 1024
#define SCALE 0.08838834764831845f   // HD^-0.5
#define NEGBIG (-1.0e30f)

typedef __attribute__((ext_vector_type(16))) __bf16 v16bf;
typedef __attribute__((ext_vector_type(8)))  float  v8f;

// A-matrix (16-bit, 16x32) element->K map for lane half (ISA 7.12.2)
__device__ __forceinline__ int akmap(int e, int half) {
    return (e & 7) + (half << 3) + ((e >> 3) << 4);
}
// B-matrix (16-bit, 32x16) element->K map: same as A (table order)
__device__ __forceinline__ int bkmap(int e, int half) {
    return (e & 7) + (half << 3) + ((e >> 3) << 4);
}

__device__ __forceinline__ v8f v8f_zero() {
    v8f z = {0.f, 0.f, 0.f, 0.f, 0.f, 0.f, 0.f, 0.f};
    return z;
}

__device__ __forceinline__ v8f wmma_bf16(v16bf a, v16bf b, v8f c) {
    return __builtin_amdgcn_wmma_f32_16x16x32_bf16(
        /*neg_a=*/false, a, /*neg_b=*/false, b,
        /*c_mod=*/(short)0, c, /*reuse_a=*/false, /*reuse_b=*/false);
}

// ---------------- fp32 -> bf16 conversion ----------------
__global__ void f32_to_bf16_kernel(const float* __restrict__ src,
                                   __bf16* __restrict__ dst, int n) {
    int i = blockIdx.x * blockDim.x + threadIdx.x;
    int stride = gridDim.x * blockDim.x;
    for (; i < n; i += stride) dst[i] = (__bf16)src[i];
}

// ---------------- generic bf16 WMMA GEMM ----------------
// C[M,N] = A[M,K] @ B[K,N]; BM=BN=128, BK=32; 256 threads (8 waves),
// wave grid 2x4, each wave computes 64x32 (4 m-frags x 2 n-frags).
__global__ __launch_bounds__(256)
void gemm_bf16_wmma(const __bf16* __restrict__ A, const __bf16* __restrict__ Bw,
                    float* __restrict__ Cf, __bf16* __restrict__ Cb,
                    int M, int N, int K, int write_f32) {
    __shared__ __bf16 As[128 * 32];   // [m][k]
    __shared__ __bf16 Bs[128 * 32];   // [n][k] (transposed at stage time)

    const int tid  = threadIdx.x;
    const int lane = tid & 31;
    const int wave = tid >> 5;
    const int half = lane >> 4;
    const int l16  = lane & 15;
    const int bm = blockIdx.y * 128;
    const int bn = blockIdx.x * 128;
    const int wm = (wave >> 2) * 64;   // 0 or 64
    const int wn = (wave & 3) * 32;    // 0,32,64,96

    v8f acc[4][2];
    #pragma unroll
    for (int fm = 0; fm < 4; ++fm)
        for (int fn = 0; fn < 2; ++fn) acc[fm][fn] = v8f_zero();

    for (int k0 = 0; k0 < K; k0 += 32) {
        __syncthreads();
        // stage A tile 128x32 (coalesced along k)
        for (int i = tid; i < 128 * 32; i += 256) {
            int r = i >> 5, c = i & 31;
            As[i] = A[(size_t)(bm + r) * K + (k0 + c)];
        }
        // stage B tile 32x128 transposed -> Bs[n][k]
        for (int i = tid; i < 32 * 128; i += 256) {
            int r = i >> 7, c = i & 127;     // r=k, c=n
            Bs[c * 32 + r] = Bw[(size_t)(k0 + r) * N + (bn + c)];
        }
        __syncthreads();

        v16bf bf[2];
        #pragma unroll
        for (int fn = 0; fn < 2; ++fn) {
            const __bf16* bp = &Bs[(wn + fn * 16 + l16) * 32];
            #pragma unroll
            for (int e = 0; e < 16; ++e) bf[fn][e] = bp[bkmap(e, half)];
        }
        #pragma unroll
        for (int fm = 0; fm < 4; ++fm) {
            v16bf af;
            const __bf16* ap = &As[(wm + fm * 16 + l16) * 32];
            #pragma unroll
            for (int e = 0; e < 16; ++e) af[e] = ap[akmap(e, half)];
            acc[fm][0] = wmma_bf16(af, bf[0], acc[fm][0]);
            acc[fm][1] = wmma_bf16(af, bf[1], acc[fm][1]);
        }
    }

    #pragma unroll
    for (int fm = 0; fm < 4; ++fm)
        #pragma unroll
        for (int fn = 0; fn < 2; ++fn)
            #pragma unroll
            for (int r = 0; r < 8; ++r) {
                int row = bm + wm + fm * 16 + half * 8 + r;
                int col = bn + wn + fn * 16 + l16;
                if (write_f32) Cf[(size_t)row * N + col] = acc[fm][fn][r];
                else           Cb[(size_t)row * N + col] = (__bf16)acc[fm][fn][r];
            }
}

// ---------------- RoPE + head-major scatter ----------------
// q_raw[b*S+s][h*HD + d] -> Qh[((b*H+h)*S+s)*HD + d] with RoPE
__global__ void rope_q_kernel(const __bf16* __restrict__ qraw,
                              const float* __restrict__ cosp,
                              const float* __restrict__ sinp,
                              __bf16* __restrict__ Qh, int total) {
    int idx = blockIdx.x * blockDim.x + threadIdx.x;
    if (idx >= total) return;
    int i = idx % 64;  int t = idx / 64;
    int s = t % SS;    t /= SS;
    int h = t % HH;
    int b = t / HH;
    size_t src = (size_t)(b * SS + s) * DD + h * HD + 2 * i;
    float x1 = (float)qraw[src];
    float x2 = (float)qraw[src + 1];
    float c  = cosp[s * 64 + i];
    float sn = sinp[s * 64 + i];
    size_t o = (size_t)((b * HH + h) * SS + s) * HD + 2 * i;
    Qh[o]     = (__bf16)(x1 * c - x2 * sn);
    Qh[o + 1] = (__bf16)(x1 * sn + x2 * c);
}

// kv_raw[b*S+s][ksel*512 + h*HD + d]; K gets RoPE, V copies.
__global__ void rope_kv_kernel(const __bf16* __restrict__ kvraw,
                               const float* __restrict__ cosp,
                               const float* __restrict__ sinp,
                               __bf16* __restrict__ Kh,
                               __bf16* __restrict__ Vh, int total) {
    int idx = blockIdx.x * blockDim.x + threadIdx.x;
    if (idx >= total) return;
    int i = idx % 64;  int t = idx / 64;
    int s = t % SS;    t /= SS;
    int h = t % KVH;
    int b = t / KVH;
    size_t base = (size_t)(b * SS + s) * NKV;
    size_t ks = base + h * HD + 2 * i;
    size_t vs = base + KVH * HD + h * HD + 2 * i;
    float x1 = (float)kvraw[ks];
    float x2 = (float)kvraw[ks + 1];
    float c  = cosp[s * 64 + i];
    float sn = sinp[s * 64 + i];
    size_t o = (size_t)((b * KVH + h) * SS + s) * HD + 2 * i;
    Kh[o]     = (__bf16)(x1 * c - x2 * sn);
    Kh[o + 1] = (__bf16)(x1 * sn + x2 * c);
    Vh[o]     = kvraw[vs];
    Vh[o + 1] = kvraw[vs + 1];
}

// ---------------- causal flash attention ----------------
// grid: (S/64, B*H), block: 128 threads = 4 waves; wave owns 16 q rows.
__global__ __launch_bounds__(128)
void attn_kernel(const __bf16* __restrict__ Q, const __bf16* __restrict__ Kh,
                 const __bf16* __restrict__ Vh, __bf16* __restrict__ Oattn) {
    __shared__ __bf16 Ks[32 * 128];
    __shared__ __bf16 Vs[32 * 128];
    __shared__ __bf16 Ps[4][16 * 32];

    const int tid  = threadIdx.x;
    const int lane = tid & 31;
    const int wave = tid >> 5;
    const int half = lane >> 4;
    const int l16  = lane & 15;

    const int bh  = blockIdx.y;
    const int b   = bh / HH;
    const int h   = bh % HH;
    const int kvh = h / GG;
    const int q0  = blockIdx.x * 64 + wave * 16;

    const __bf16* Qp = Q  + (size_t)(b * HH + h) * SS * HD;
    const __bf16* Kp = Kh + (size_t)(b * KVH + kvh) * SS * HD;
    const __bf16* Vp = Vh + (size_t)(b * KVH + kvh) * SS * HD;

    // Q fragments: 16 rows x 128 dims = 4 A-frags (K-steps of 32)
    v16bf qf[4];
    #pragma unroll
    for (int kk = 0; kk < 4; ++kk) {
        const __bf16* qp = Qp + (size_t)(q0 + l16) * HD + kk * 32;
        #pragma unroll
        for (int e = 0; e < 16; ++e) qf[kk][e] = qp[akmap(e, half)];
    }

    v8f acc[8];
    #pragma unroll
    for (int fn = 0; fn < 8; ++fn) acc[fn] = v8f_zero();
    float mrow[8], lrow[8];
    #pragma unroll
    for (int r = 0; r < 8; ++r) { mrow[r] = NEGBIG; lrow[r] = 0.f; }

    const int ntiles = (blockIdx.x + 1) * 2;   // 32 keys per tile, causal bound
    for (int kt = 0; kt < ntiles; ++kt) {
        __syncthreads();
        for (int i = tid; i < 32 * 128; i += 128) {
            int r = i >> 7, c = i & 127;
            size_t g = (size_t)(kt * 32 + r) * HD + c;
            Ks[i] = Kp[g];
            Vs[i] = Vp[g];
        }
        __syncthreads();
        if (kt * 32 > q0 + 15) continue;   // tile fully masked for this wave

        // scores 16x32: two 16x16 C tiles
        v8f s0 = v8f_zero(), s1 = v8f_zero();
        #pragma unroll
        for (int kk = 0; kk < 4; ++kk) {
            v16bf b0, b1;
            const __bf16* k0p = &Ks[(size_t)l16 * 128 + kk * 32];
            const __bf16* k1p = &Ks[(size_t)(16 + l16) * 128 + kk * 32];
            #pragma unroll
            for (int e = 0; e < 16; ++e) {
                b0[e] = k0p[bkmap(e, half)];
                b1[e] = k1p[bkmap(e, half)];
            }
            s0 = wmma_bf16(qf[kk], b0, s0);
            s1 = wmma_bf16(qf[kk], b1, s1);
        }

        // online softmax (row = q0 + 8*half + r; cols striped over 16 lanes)
        #pragma unroll
        for (int r = 0; r < 8; ++r) {
            int row  = q0 + half * 8 + r;
            int key0 = kt * 32 + l16;
            int key1 = kt * 32 + 16 + l16;
            float v0 = s0[r] * SCALE; if (key0 > row) v0 = NEGBIG;
            float v1 = s1[r] * SCALE; if (key1 > row) v1 = NEGBIG;
            float tm = fmaxf(v0, v1);
            #pragma unroll
            for (int off = 1; off < 16; off <<= 1)
                tm = fmaxf(tm, __shfl_xor(tm, off, 32));
            float mnew = fmaxf(mrow[r], tm);
            float p0 = __expf(v0 - mnew);
            float p1 = __expf(v1 - mnew);
            float ps = p0 + p1;
            #pragma unroll
            for (int off = 1; off < 16; off <<= 1)
                ps += __shfl_xor(ps, off, 32);
            float corr = __expf(mrow[r] - mnew);
            lrow[r] = lrow[r] * corr + ps;
            mrow[r] = mnew;
            #pragma unroll
            for (int fn = 0; fn < 8; ++fn) acc[fn][r] *= corr;
            Ps[wave][(half * 8 + r) * 32 + l16]      = (__bf16)p0;
            Ps[wave][(half * 8 + r) * 32 + 16 + l16] = (__bf16)p1;
        }
        // wave-private LDS transpose: drain DS writes before re-reading
        asm volatile("s_wait_dscnt 0" ::: "memory");

        v16bf pf;
        const __bf16* pp = &Ps[wave][l16 * 32];
        #pragma unroll
        for (int e = 0; e < 16; ++e) pf[e] = pp[akmap(e, half)];

        #pragma unroll
        for (int fn = 0; fn < 8; ++fn) {
            v16bf vb;
            #pragma unroll
            for (int e = 0; e < 16; ++e)
                vb[e] = Vs[(size_t)bkmap(e, half) * 128 + fn * 16 + l16];
            acc[fn] = wmma_bf16(pf, vb, acc[fn]);
        }
    }

    #pragma unroll
    for (int r = 0; r < 8; ++r) {
        float inv = 1.f / lrow[r];
        int srow = q0 + half * 8 + r;
        #pragma unroll
        for (int fn = 0; fn < 8; ++fn)
            Oattn[(size_t)(b * SS + srow) * DD + h * HD + fn * 16 + l16] =
                (__bf16)(acc[fn][r] * inv);
    }
}

// ---------------- host launch ----------------
extern "C" void kernel_launch(void* const* d_in, const int* in_sizes, int n_in,
                              void* d_out, int out_size, void* d_ws, size_t ws_size,
                              hipStream_t stream) {
    (void)in_sizes; (void)n_in; (void)out_size; (void)ws_size;
    const float* x    = (const float*)d_in[0];
    // d_in[1] = mask (unused; causal mask applied analytically)
    const float* cosp = (const float*)d_in[2];
    const float* sinp = (const float*)d_in[3];
    const float* Wq   = (const float*)d_in[4];
    const float* Wkv  = (const float*)d_in[5];
    const float* Wo   = (const float*)d_in[6];
    float* out = (float*)d_out;

    const int M = BB * SS;                    // 4096
    char* ws = (char*)d_ws;
    size_t off = 0;
    auto alloc_bf = [&](size_t nelem) {
        __bf16* p = (__bf16*)(ws + off);
        off += nelem * sizeof(__bf16);
        return p;
    };
    __bf16* x_bf   = alloc_bf((size_t)M * DD);       // also reused as Qh later
    __bf16* Wq_bf  = alloc_bf((size_t)DD * DD);
    __bf16* Wkv_bf = alloc_bf((size_t)DD * NKV);
    __bf16* Wo_bf  = alloc_bf((size_t)DD * DD);
    __bf16* q_raw  = alloc_bf((size_t)M * DD);       // also reused as attn out
    __bf16* kv_raw = alloc_bf((size_t)M * NKV);
    __bf16* Kh     = alloc_bf((size_t)BB * KVH * SS * HD);
    __bf16* Vh     = alloc_bf((size_t)BB * KVH * SS * HD);
    __bf16* Qh     = x_bf;     // x_bf dead after the two projection GEMMs
    __bf16* Oattn  = q_raw;    // q_raw dead after rope_q

    // 1) convert inputs to bf16
    {
        int n;
        n = M * DD;    f32_to_bf16_kernel<<<(n + 255) / 256, 256, 0, stream>>>(x,   x_bf,   n);
        n = DD * DD;   f32_to_bf16_kernel<<<(n + 255) / 256, 256, 0, stream>>>(Wq,  Wq_bf,  n);
        n = DD * NKV;  f32_to_bf16_kernel<<<(n + 255) / 256, 256, 0, stream>>>(Wkv, Wkv_bf, n);
        n = DD * DD;   f32_to_bf16_kernel<<<(n + 255) / 256, 256, 0, stream>>>(Wo,  Wo_bf,  n);
    }
    // 2) projections (WMMA GEMMs) -> bf16 scratch
    gemm_bf16_wmma<<<dim3(DD / 128, M / 128), 256, 0, stream>>>(
        x_bf, Wq_bf, nullptr, q_raw, M, DD, DD, 0);
    gemm_bf16_wmma<<<dim3(NKV / 128, M / 128), 256, 0, stream>>>(
        x_bf, Wkv_bf, nullptr, kv_raw, M, NKV, DD, 0);
    // 3) RoPE + head-major scatter
    {
        int tq = BB * SS * HH * 64;
        rope_q_kernel<<<(tq + 255) / 256, 256, 0, stream>>>(q_raw, cosp, sinp, Qh, tq);
        int tk = BB * SS * KVH * 64;
        rope_kv_kernel<<<(tk + 255) / 256, 256, 0, stream>>>(kv_raw, cosp, sinp, Kh, Vh, tk);
    }
    // 4) causal flash attention (WMMA for QK^T and PV)
    attn_kernel<<<dim3(SS / 64, BB * HH), 128, 0, stream>>>(Qh, Kh, Vh, Oattn);
    // 5) output projection -> f32 d_out
    gemm_bf16_wmma<<<dim3(DD / 128, M / 128), 256, 0, stream>>>(
        Oattn, Wo_bf, out, nullptr, M, DD, DD, 1);
}
